// NCELoss_39788577030814
// MI455X (gfx1250) — matrix operand order repaired and checked by reading, not compiled, not run on hardware
//
#include <hip/hip_runtime.h>
#include <hip/hip_bf16.h>

// ---------------- types ----------------
typedef __bf16 v16bf __attribute__((ext_vector_type(16)));
typedef __bf16 v8bf  __attribute__((ext_vector_type(8)));
typedef __bf16 v4bf  __attribute__((ext_vector_type(4)));
typedef float  v8f   __attribute__((ext_vector_type(8)));

#define B_    256
#define T_    128
#define DIN   1024
#define DH    2048
#define DOUT  256
#define MROWS (B_ * T_)      // 32768
#define MT    64             // feats rows per workgroup
#define XSTR  1032           // padded LDS stride for x tile (bf16 elems)
#define HSTR  72             // padded LDS stride for H tile (bf16 elems)

#define W1T_BYTES  (size_t)(DIN * DH * 2)          //  4 MB
#define W2T_BYTES  (size_t)(DH * DOUT * 2)         //  1 MB
#define FEATS_BYTES (size_t)(MROWS * DOUT * 4)     // 32 MB

__device__ __forceinline__ v16bf cat16(v8bf lo, v8bf hi) {
  return __builtin_shufflevector(lo, hi, 0,1,2,3,4,5,6,7,8,9,10,11,12,13,14,15);
}
__device__ __forceinline__ v8f wmma_bf16(v16bf a, v16bf b, v8f c) {
  return __builtin_amdgcn_wmma_f32_16x16x32_bf16(false, a, false, b, (short)0, c, false, false);
}
__device__ __forceinline__ v8f zero8() {
  v8f z;
#pragma unroll
  for (int i = 0; i < 8; ++i) z[i] = 0.f;
  return z;
}
// A-fragment: 16-bit 16x32, lane<16 holds K = base..base+15 contiguous (K offset 16*(lane>=16) added by caller)
__device__ __forceinline__ v16bf ldA(const __bf16* rowp, int kbase) {
  v8bf lo = *(const v8bf*)(rowp + kbase);
  v8bf hi = *(const v8bf*)(rowp + kbase + 8);
  return cat16(lo, hi);
}
// B-fragment: lane<16 holds K = base..base+15 contiguous (K offset 16*(lane>=16) added by caller)
__device__ __forceinline__ v16bf ldB(const __bf16* rowp, int kbase) {
  v8bf lo = *(const v8bf*)(rowp + kbase);
  v8bf hi = *(const v8bf*)(rowp + kbase + 8);
  return cat16(lo, hi);
}

// ---------------- kernel 1: weight transpose + bf16 convert ----------------
__global__ __launch_bounds__(256)
void prep_kernel(const float* __restrict__ W1, const float* __restrict__ W2,
                 __bf16* __restrict__ W1T, __bf16* __restrict__ W2T) {
  int tid = blockIdx.x * blockDim.x + threadIdx.x;
  int stride = gridDim.x * blockDim.x;
  for (int id = tid; id < DIN * DH; id += stride) {      // coalesced read of W1
    int d = id / DH, h = id % DH;
    W1T[(size_t)h * DIN + d] = (__bf16)W1[id];
  }
  for (int id = tid; id < DH * DOUT; id += stride) {     // coalesced read of W2
    int h = id / DOUT, c = id % DOUT;
    W2T[(size_t)c * DH + h] = (__bf16)W2[id];
  }
}

// ---------------- kernel 2: fused MLP, feats = relu(x@W1+b1)@W2+b2 ----------------
__global__ __launch_bounds__(256)
void mlp_kernel(const float* __restrict__ x, const float* __restrict__ b1,
                const float* __restrict__ b2,
                const __bf16* __restrict__ W1T, const __bf16* __restrict__ W2T,
                float* __restrict__ feats) {
  __shared__ __bf16 lx[MT * XSTR];   // 129 KB: x tile in bf16
  __shared__ __bf16 lh[MT * HSTR];   //   9 KB: hidden chunk (64 wide)

  const int tid  = threadIdx.x;
  const int wave = tid >> 5;
  const int lane = tid & 31;
  const int l16  = lane & 15;
  const int lh16 = lane >> 4;                 // 0 or 1
  const int rowbase = blockIdx.x * MT;

  // ---- stage x tile: global fp32 -> LDS bf16 (x read from HBM exactly once) ----
  {
    const float4* x4 = (const float4*)x;
#pragma unroll 4
    for (int it = 0; it < 64; ++it) {
      int id = it * 256 + tid;
      int r  = id >> 8;          // 256 float4 per row
      int c4 = id & 255;
      float4 v = x4[(size_t)(rowbase + r) * 256 + c4];
      v4bf bb;
      bb[0] = (__bf16)v.x; bb[1] = (__bf16)v.y;
      bb[2] = (__bf16)v.z; bb[3] = (__bf16)v.w;
      *(v4bf*)(lx + r * XSTR + c4 * 4) = bb;
    }
  }
  __syncthreads();

  // work mapping: 8 waves = 4 (M groups) x 2 (N halves)
  const int m1 = (wave & 3) * 16;             // phase-1/2 M base within tile
  const int np = wave >> 2;                   // phase-1 n-pair (32 H cols)
  const int n2 = (wave >> 2) * 128;           // phase-2 C col base (8 subtiles)

  v8f acc2[8];
#pragma unroll
  for (int j = 0; j < 8; ++j) acc2[j] = zero8();

  const __bf16* lxA = lx + (m1 + l16) * XSTR + lh16 * 16;
  const __bf16* lhA = lh + (m1 + l16) * HSTR + lh16 * 16;

  for (int ch = 0; ch < DH / 64; ++ch) {
    // ---- phase 1: H[64x64] = relu(xtile @ W1chunk + b1) ----
    v8f c0 = zero8(), c1 = zero8();
    const int h0 = ch * 64 + np * 32;
    const __bf16* w1r0 = W1T + (size_t)(h0 + l16)      * DIN + lh16 * 16;
    const __bf16* w1r1 = W1T + (size_t)(h0 + 16 + l16) * DIN + lh16 * 16;
#pragma unroll 4
    for (int ks = 0; ks < DIN / 32; ++ks) {
      int ko = ks * 32;
      v16bf a  = ldA(lxA, ko);
      v16bf b0 = ldB(w1r0, ko);
      v16bf b1f = ldB(w1r1, ko);
      c0 = wmma_bf16(a, b0, c0);
      c1 = wmma_bf16(a, b1f, c1);
    }
    __syncthreads();   // prior phase-2 readers of lh are done
    {
      float bv0 = b1[h0 + l16];
      float bv1 = b1[h0 + 16 + l16];
      int nl = np * 32 + l16;
#pragma unroll
      for (int r = 0; r < 8; ++r) {
        int m = m1 + r + 8 * lh16;            // C layout: lanes>=16 hold M=8+r
        float v0 = c0[r] + bv0; v0 = v0 > 0.f ? v0 : 0.f;
        float v1 = c1[r] + bv1; v1 = v1 > 0.f ? v1 : 0.f;
        lh[m * HSTR + nl]      = (__bf16)v0;
        lh[m * HSTR + nl + 16] = (__bf16)v1;
      }
    }
    __syncthreads();

    // ---- phase 2: acc += H @ W2chunk ----
#pragma unroll
    for (int s = 0; s < 2; ++s) {
      v16bf a = ldA(lhA, s * 32);
#pragma unroll
      for (int j = 0; j < 8; ++j) {
        int cc = n2 + j * 16 + l16;
        const __bf16* w2r = W2T + (size_t)cc * DH + ch * 64 + s * 32 + lh16 * 16;
        v16bf b = ldB(w2r, 0);
        acc2[j] = wmma_bf16(a, b, acc2[j]);
      }
    }
  }

  // ---- epilogue: + b2, store feats ----
#pragma unroll
  for (int j = 0; j < 8; ++j) {
    int cc = n2 + j * 16 + l16;
    float bias = b2[cc];
#pragma unroll
    for (int r = 0; r < 8; ++r) {
      int m = m1 + r + 8 * lh16;
      feats[(size_t)(rowbase + m) * DOUT + cc] = acc2[j][r] + bias;
    }
  }
}

// ---------------- kernel 3: per-anchor contrastive loss row ----------------
__global__ __launch_bounds__(256)
void loss_kernel(const float* __restrict__ feats, const int* __restrict__ label,
                 const int* __restrict__ idx, float* __restrict__ lossmat) {
  __shared__ float sa[256];
  __shared__ float sred[16];

  const int i = blockIdx.x;
  const int c = threadIdx.x;
  const int wave = c >> 5, lane = c & 31;
  const int t  = idx[i];
  const int li = label[i];

  const float a = feats[((size_t)i * T_ + t) * DOUT + c];
  sa[c] = a;

  // anchor norm (block reduction)
  float asq = a * a;
#pragma unroll
  for (int off = 16; off; off >>= 1) asq += __shfl_xor(asq, off, 32);
  if (lane == 0) sred[wave] = asq;
  __syncthreads();
  float an2 = 0.f;
#pragma unroll
  for (int w = 0; w < 8; ++w) an2 += sred[w];
  const float an = sqrtf(an2);

  // pass 1 (channel-parallel): positive stats
  float pos_sum = 0.f, pos_sq = 0.f;
  int P = 0;
  for (int k = 0; k < B_; ++k) {
    float g = feats[((size_t)k * T_ + t) * DOUT + c];
    bool isp = (label[k] == li) && (k != i);
    if (isp) { pos_sum += g; pos_sq += g * g; ++P; }
  }

  // pass 2 (k-parallel by waves): negative mean of cosine similarities
  float negsum = 0.f, negcnt = 0.f;
  for (int kk = 0; kk < B_ / 8; ++kk) {
    int k = kk * 8 + wave;
    const float* gr = feats + ((size_t)k * T_ + t) * DOUT + lane * 8;
    float d = 0.f, q = 0.f;
#pragma unroll
    for (int u = 0; u < 8; ++u) {
      float g = gr[u];
      d += sa[lane * 8 + u] * g;
      q += g * g;
    }
#pragma unroll
    for (int off = 16; off; off >>= 1) {
      d += __shfl_xor(d, off, 32);
      q += __shfl_xor(q, off, 32);
    }
    if (label[k] != li) {
      float den = an * sqrtf(q);
      den = den < 1e-8f ? 1e-8f : den;
      negsum += d / den;
      negcnt += 1.f;
    }
  }
  __syncthreads();
  if (lane == 0) { sred[wave] = negsum; sred[8 + wave] = negcnt; }
  __syncthreads();
  float ns = 0.f, nc = 0.f;
#pragma unroll
  for (int w = 0; w < 8; ++w) { ns += sred[w]; nc += sred[8 + w]; }
  const float neg_mean = ns / fmaxf(nc, 1.f);

  float denom = sqrtf((float)P) * fabsf(a) * sqrtf(pos_sq);
  denom = fmaxf(denom, 1e-8f);
  float pos_sim = a * pos_sum / denom;
  float loss = 0.2f - pos_sim + neg_mean;
  loss = loss > 0.f ? loss : 0.f;
  loss = (P > 0) ? loss : 0.f;
  lossmat[(size_t)i * DOUT + c] = loss;
}

// ---------------- kernel 4: deterministic reduction over anchors ----------------
__global__ __launch_bounds__(256)
void reduce_kernel(const float* __restrict__ lossmat, float* __restrict__ out) {
  int c = threadIdx.x;
  float s = 0.f;
  for (int i = 0; i < B_; ++i) s += lossmat[(size_t)i * DOUT + c];
  out[c] = s;
}

// ---------------- launch ----------------
extern "C" void kernel_launch(void* const* d_in, const int* in_sizes, int n_in,
                              void* d_out, int out_size, void* d_ws, size_t ws_size,
                              hipStream_t stream) {
  const float* x     = (const float*)d_in[0];
  const int*   label = (const int*)d_in[1];
  const int*   idx   = (const int*)d_in[2];
  const float* W1    = (const float*)d_in[3];
  const float* b1    = (const float*)d_in[4];
  const float* W2    = (const float*)d_in[5];
  const float* b2    = (const float*)d_in[6];
  float* out = (float*)d_out;

  char* ws = (char*)d_ws;
  __bf16* W1T   = (__bf16*)ws;
  __bf16* W2T   = (__bf16*)(ws + W1T_BYTES);
  float*  feats = (float*)(ws + W1T_BYTES + W2T_BYTES);
  float*  lossm = (float*)(ws + W1T_BYTES + W2T_BYTES + FEATS_BYTES);

  hipLaunchKernelGGL(prep_kernel, dim3(512), dim3(256), 0, stream, W1, W2, W1T, W2T);
  hipLaunchKernelGGL(mlp_kernel,  dim3(MROWS / MT), dim3(256), 0, stream,
                     x, b1, b2, W1T, W2T, feats);
  hipLaunchKernelGGL(loss_kernel, dim3(B_), dim3(256), 0, stream,
                     feats, label, idx, lossm);
  hipLaunchKernelGGL(reduce_kernel, dim3(1), dim3(256), 0, stream, lossm, out);
}
